// UpDownLayer_274877907667
// MI455X (gfx1250) — matrix operand order repaired and checked by reading, not compiled, not run on hardware
//
#include <hip/hip_runtime.h>

// Problem constants (from reference): x(1,2,196608,32) f32, adjc(196608,9) int,
// W(9,32,4,64) f32, b(4,64) f32. out (1,2,196608,4,64) f32.
#define T_DIM    2
#define N_NODES  196608
#define F_IN     32
#define K_NB     9
#define COLS     256                        // UP*OUT_FEATURES = 4*64
#define M_TOTAL  (T_DIM * N_NODES)          // 393216 GEMM rows
#define TILE_M   32                         // rows per workgroup tile
#define NUM_TILES (M_TOTAL / TILE_M)        // 12288
#define LDSW_HALVES (K_NB * COLS * F_IN)    // 73728 f16
#define LDSW_BYTES  (LDSW_HALVES * 2)       // 147456 B of W in LDS
#define ABUF_BYTES  (TILE_M * K_NB * F_IN * 2)  // 18432 B per A stage buffer
#define LDS_TOTAL   (LDSW_BYTES + 2 * ABUF_BYTES) // 184320 B
#define ACHUNKS     (ABUF_BYTES / 16)       // 1152 16-byte async chunks per tile
#define XH_HALVES   ((size_t)M_TOTAL * F_IN) // 12,582,912 f16 in d_ws

typedef __attribute__((ext_vector_type(16))) _Float16 v16h;
typedef __attribute__((ext_vector_type(8)))  _Float16 v8h;
typedef __attribute__((ext_vector_type(8)))  float    v8f;

// ---- Prep 1: convert x (f32) -> xh (f16) in workspace. 8 elems/thread. ----
__global__ void __launch_bounds__(256) cvt_x_f16(const float* __restrict__ x,
                                                 _Float16* __restrict__ xh) {
  size_t i = ((size_t)blockIdx.x * 256 + threadIdx.x) * 8;
  float4 a = *(const float4*)(x + i);
  float4 b = *(const float4*)(x + i + 4);
  v8h h;
  h[0] = (_Float16)a.x; h[1] = (_Float16)a.y; h[2] = (_Float16)a.z; h[3] = (_Float16)a.w;
  h[4] = (_Float16)b.x; h[5] = (_Float16)b.y; h[6] = (_Float16)b.z; h[7] = (_Float16)b.w;
  *(v8h*)(xh + i) = h;
}

// ---- Prep 2: W[k][f][u][o] f32 -> wh[k][col=u*64+o][f] f16 (B-matrix layout) ----
__global__ void __launch_bounds__(256) cvt_w_f16(const float* __restrict__ W,
                                                 _Float16* __restrict__ wh) {
  int e   = blockIdx.x * 256 + threadIdx.x;   // 0 .. 73727
  int f   = e & 31;
  int col = (e >> 5) & 255;
  int k   = e >> 13;
  int u   = col >> 6;
  int o   = col & 63;
  wh[e] = (_Float16)W[(((k * F_IN + f) * 4 + u) << 6) + o];
}

// Async-gather one 32-row tile of neighbor features into an LDS A-buffer.
// Buffer layout: [seg = local_row*9 + k][64 B of f16 features].
// 1152 16B chunks, 5 async ops per thread (tail clamped: duplicate writes of
// identical data to the same LDS address are benign).
__device__ __forceinline__ void stage_tile(int tid, int tile, unsigned ldsbase,
                                           const int* __restrict__ adjc,
                                           const unsigned char* __restrict__ xb) {
#pragma unroll
  for (int i = 0; i < 5; ++i) {
    unsigned c = (unsigned)tid + (unsigned)i * 256u;
    c = (c > (unsigned)(ACHUNKS - 1)) ? (unsigned)(ACHUNKS - 1) : c;
    unsigned seg  = c >> 2;                       // 0..287 = row*9 + k
    unsigned part = (c & 3u) * 16u;               // byte within 64B segment
    unsigned row  = seg / 9u;                     // local row 0..31
    unsigned k    = seg - row * 9u;               // neighbor 0..8
    unsigned gr   = (unsigned)tile * TILE_M + row;
    unsigned tt   = (gr >= (unsigned)N_NODES) ? 1u : 0u;
    unsigned nn   = gr - tt * (unsigned)N_NODES;
    int nb = adjc[nn * K_NB + k];
    unsigned goff = (tt * (unsigned)N_NODES + (unsigned)nb) * (F_IN * 2u) + part;
    unsigned l    = ldsbase + seg * 64u + part;
    unsigned long long g = (unsigned long long)xb + goff;
    asm volatile("global_load_async_to_lds_b128 %0, %1, off"
                 :: "v"(l), "v"(g) : "memory");
  }
}

// ---- Main: async-pipelined gather + GEMM via V_WMMA_F32_16X16X32_F16 ----
__global__ void __launch_bounds__(256) gather_gemm_wmma(
    const _Float16* __restrict__ xh,   // [T][N][F] f16
    const int*      __restrict__ adjc, // [N][K]
    const _Float16* __restrict__ wh,   // [K][COLS][F] f16
    const float*    __restrict__ bias, // [COLS]
    float*          __restrict__ out)  // [M_TOTAL][COLS]
{
  extern __shared__ __align__(128) unsigned char lds[];  // W | Abuf0 | Abuf1

  const int tid = threadIdx.x;
  const unsigned char* xb = (const unsigned char*)xh;

  // Stage full W (f16, B-layout) into LDS via the async-to-LDS path.
  {
    const unsigned      lbase = (unsigned)tid * 16u;
    const unsigned long long gbase =
        (unsigned long long)((const unsigned char*)wh) + lbase;
#pragma unroll
    for (int i = 0; i < 36; ++i) {
      unsigned           l = lbase + (unsigned)i * 4096u;
      unsigned long long g = gbase + (unsigned long long)i * 4096ull;
      asm volatile("global_load_async_to_lds_b128 %0, %1, off"
                   :: "v"(l), "v"(g) : "memory");
    }
    asm volatile("s_wait_asynccnt 0x0" ::: "memory");
  }
  __syncthreads();

  const int lane    = tid & 31;
  const int wave    = tid >> 5;
  const int m_sub   = wave & 1;        // which 16-row subtile of the 32-row tile
  const int n_quad  = wave >> 1;       // which 64-column quad
  const int l15     = lane & 15;
  const int hi      = lane >> 4;       // lane group 0/1
  const int fbaseA  = hi * 16;         // A-frag K offsets: contiguous k = 16*hi + s
  const int fbaseB  = hi * 16;         // B-frag K offsets: contiguous k = 16*hi + s
  const int colbase = n_quad * 64;
  const int m_loc   = m_sub * 16 + l15;  // local row this lane's A-frag covers

  // Pin all 36 B-fragments in VGPRs for the lifetime of the block.
  v16h bfr[K_NB][4];
#pragma unroll
  for (int k = 0; k < K_NB; ++k)
#pragma unroll
    for (int j = 0; j < 4; ++j)
      bfr[k][j] = *(const v16h*)(lds +
          (size_t)((k * COLS + colbase + j * 16 + l15) * F_IN) * 2 + fbaseB * 2);

  float bj[4];
#pragma unroll
  for (int j = 0; j < 4; ++j) bj[j] = bias[colbase + j * 16 + l15];

  unsigned char* ob = (unsigned char*)out;
  const int stride = (int)gridDim.x;

  // Prologue: stage the first tile into buffer 0.
  int tile0 = blockIdx.x;
  if (tile0 < NUM_TILES) stage_tile(tid, tile0, LDSW_BYTES, adjc, xb);

  int buf = 0;
  for (int tile = tile0; tile < NUM_TILES; tile += stride, buf ^= 1) {
    // 1) Current buffer's async gathers (issued last iteration) complete.
    asm volatile("s_wait_asynccnt 0x0" ::: "memory");
    // 2) One barrier: all waves' staging done AND previous iteration's LDS
    //    reads finished before anyone overwrites a buffer.
    __syncthreads();

    // 3) Read this tile's A-fragments from LDS into registers (cheap ds hits).
    const unsigned abase = LDSW_BYTES + (unsigned)buf * ABUF_BYTES;
    v8h alo[K_NB], ahi[K_NB];
#pragma unroll
    for (int k = 0; k < K_NB; ++k) {
      const unsigned o = abase + (unsigned)(m_loc * K_NB + k) * 64u
                       + (unsigned)fbaseA * 2u;
      alo[k] = *(const v8h*)(lds + o);        // K = fbaseA .. +7
      ahi[k] = *(const v8h*)(lds + o + 16u);  // K = fbaseA+8 .. +15
    }

    // 4) Kick off the NEXT tile's gather (adjc loads + async issue); its
    //    latency is hidden behind this tile's 36 WMMAs.
    const int tnext = tile + stride;
    if (tnext < NUM_TILES)
      stage_tile(tid, tnext, LDSW_BYTES + (unsigned)(buf ^ 1) * ABUF_BYTES,
                 adjc, xb);

    // 5) Compute.
    v8f acc[4] = {};
#pragma unroll
    for (int k = 0; k < K_NB; ++k) {
      v16h a;
#pragma unroll
      for (int i = 0; i < 8; ++i) { a[i] = alo[k][i]; a[8 + i] = ahi[k][i]; }
#pragma unroll
      for (int j = 0; j < 4; ++j) {
        acc[j] = __builtin_amdgcn_wmma_f32_16x16x32_f16(
            /*neg_a=*/false, a, /*neg_b=*/false, bfr[k][j],
            /*c_mod=*/(short)0, acc[j], /*reuse_a=*/false, /*reuse_b=*/false);
      }
    }

    // 6) Store: C/D layout -> lanes 0-15 rows v, lanes 16-31 rows v+8.
    const unsigned rowbase =
        (unsigned)(tile * TILE_M + m_sub * 16) + (hi ? 8u : 0u);
#pragma unroll
    for (int j = 0; j < 4; ++j) {
      const unsigned off =
          (rowbase * COLS + (unsigned)(colbase + j * 16 + l15)) * 4u;
#pragma unroll
      for (int v = 0; v < 8; ++v)
        *(float*)(ob + off + (unsigned)v * (COLS * 4u)) = acc[j][v] + bj[j];
    }
  }
}

extern "C" void kernel_launch(void* const* d_in, const int* in_sizes, int n_in,
                              void* d_out, int out_size, void* d_ws, size_t ws_size,
                              hipStream_t stream) {
  const float* x    = (const float*)d_in[0];
  const int*   adjc = (const int*)d_in[1];
  const float* W    = (const float*)d_in[2];
  const float* b    = (const float*)d_in[3];
  float*       out  = (float*)d_out;

  _Float16* xh = (_Float16*)d_ws;                                        // 25,165,824 B
  _Float16* wh = (_Float16*)((char*)d_ws + XH_HALVES * sizeof(_Float16)); // +147,456 B

  // Stream-ordered: converts complete before the GEMM consumes them.
  cvt_x_f16<<<(int)(XH_HALVES / 2048), 256, 0, stream>>>(x, xh);
  cvt_w_f16<<<LDSW_HALVES / 256, 256, 0, stream>>>(W, wh);
  gather_gemm_wmma<<<1536, 256, LDS_TOTAL, stream>>>(xh, adjc, wh, b, out);
}
